// NT_XentLoss_57836029608255
// MI455X (gfx1250) — matrix operand order repaired and checked by reading, not compiled, not run on hardware
//
#include <hip/hip_runtime.h>
#include <hip/hip_bf16.h>

// ---------------------------------------------------------------------------
// NT-Xent (SimCLR) loss, fused: normalize -> f16 WMMA GEMM (never materialize
// the 8192x8192 sim matrix) -> online sum(exp) per row -> logsumexp - pos.
// Logits are bounded in [-2, 2] (cosine / 0.5) so no running-max is needed.
//
// B panels staged in LDS via CDNA5 async DMA (global_load_async_to_lds_b128),
// TRIPLE-buffered with a 2-tile prefetch lead, shared by all 8 waves of the
// workgroup (8x L2 traffic cut). Inside a tile, all 8 B fragments are pulled
// from LDS into registers first, then 8 WMMAs run back-to-back (single
// s_wait_dscnt per tile instead of one per k-step).
// ---------------------------------------------------------------------------

typedef __attribute__((ext_vector_type(16))) _Float16 v16h;
typedef __attribute__((ext_vector_type(8)))  _Float16 v8h;
typedef __attribute__((ext_vector_type(8)))  float    v8f;

#define D        256
#define NPAIR    4096
#define TWON     8192
#define COLSLICE 1024           // columns per grid.y slice
#define NTILES   (COLSLICE/16)  // 64 column tiles per slice
#define INV_T    2.0f           // 1 / TEMPERATURE
#define KSTRIDE  24             // halves per LDS panel k-row (48B, bank-staggered)

// --- kernel 0: zero the accumulators (harness poisons ws/out with 0xAA) ----
__global__ void nt_init(float* S, float* P, float* out) {
    int idx = blockIdx.x * blockDim.x + threadIdx.x;
    if (idx < TWON) { S[idx] = 0.0f; P[idx] = 0.0f; }
    if (idx == 0)   { out[0] = 0.0f; }
}

// --- kernel 1: row-normalize, emit f16 row-major + f16 transposed ----------
__global__ void nt_normalize(const float* __restrict__ zi,
                             const float* __restrict__ zj,
                             _Float16* __restrict__ zn_rm,   // [TWON][D]
                             _Float16* __restrict__ zn_tr) { // [D][TWON]
    const int row = blockIdx.x;
    const int t   = threadIdx.x;
    const float* src = (row < NPAIR) ? (zi + (size_t)row * D)
                                     : (zj + (size_t)(row - NPAIR) * D);
    float v = src[t];

    __shared__ float red[D];
    red[t] = v * v;
    __syncthreads();
    #pragma unroll
    for (int s = D / 2; s > 0; s >>= 1) {
        if (t < s) red[t] += red[t + s];
        __syncthreads();
    }
    const float inv = 1.0f / fmaxf(sqrtf(red[0]), 1e-8f);
    const _Float16 h = (_Float16)(v * inv);
    zn_rm[(size_t)row * D + t]    = h;
    zn_tr[(size_t)t * TWON + row] = h;
}

// Issue one k-row (32 bytes) of the B panel: global -> LDS, async (ASYNCcnt).
// INST_OFFSET is added to BOTH the LDS and global addresses per the ISA.
__device__ __forceinline__ void panel_fill_async(const _Float16* gsrc,
                                                 unsigned lds_addr) {
    unsigned long long ga = (unsigned long long)(uintptr_t)gsrc;
    asm volatile("global_load_async_to_lds_b128 %0, %1, off"
                 :: "v"(lds_addr), "v"(ga) : "memory");
    asm volatile("global_load_async_to_lds_b128 %0, %1, off offset:16"
                 :: "v"(lds_addr), "v"(ga) : "memory");
}

__device__ __forceinline__ v16h load_bfrag(const _Float16* p) {
    v8h b0 = *(const v8h*)(p);
    v8h b1 = *(const v8h*)(p + 8);
    v16h b;
    #pragma unroll
    for (int e = 0; e < 8; ++e) { b[e] = b0[e]; b[e + 8] = b1[e]; }
    return b;
}

// --- kernel 2: fused sim GEMM + masked sum(exp) + positive-logit capture ---
// block = 256 threads = 8 waves; wave w owns row tile R = (bx*8+w)*16.
// grid.y slices the 8192 columns into 8 chunks of 1024.
__global__ void
__launch_bounds__(256, 2)
nt_simloss(const _Float16* __restrict__ zn_rm,
           const _Float16* __restrict__ zn_tr,
           float* __restrict__ S,     // per-row sum of exp(logit), diag masked
           float* __restrict__ P) {   // per-row positive logit
    const int tid  = threadIdx.x;     // 0..255, doubles as the panel k-row id
    const int lane = tid & 31;
    const int wave = tid >> 5;
    const int m    = lane & 15;       // A: M index / B: N index / C: N index
    const int hi   = lane >> 4;       // upper half-wave flag
    const int R    = (blockIdx.x * 8 + wave) * 16;
    const int col0 = blockIdx.y * COLSLICE;

    // triple-buffered B panel: [buf][k-row 0..255][16 cols], rows padded 48B
    __shared__ _Float16 panel[3][D * KSTRIDE];

    // ---- A fragments: rows [R, R+16), full K=256 (8 x 16x16x32 chunks) ----
    // ISA layout (16-bit A 16x32): lane<16 holds K {0..7,16..23} of row m,
    // lane>=16 holds K {8..15,24..31}. All contiguous 16B loads.
    v16h a[8];
    const _Float16* arow = zn_rm + (size_t)(R + m) * D + hi * 8;
    #pragma unroll
    for (int kk = 0; kk < 8; ++kk) {
        v8h lo  = *(const v8h*)(arow + kk * 32);
        v8h hi8 = *(const v8h*)(arow + kk * 32 + 16);
        v16h f;
        #pragma unroll
        for (int e = 0; e < 8; ++e) { f[e] = lo[e]; f[e + 8] = hi8[e]; }
        a[kk] = f;
    }

    float fsum[8], fpos[8];
    #pragma unroll
    for (int v = 0; v < 8; ++v) { fsum[v] = 0.0f; fpos[v] = 0.0f; }

    const int krow = m + 16 * hi;     // equals lane; unused

    // panel fill: thread t owns k-row t (32B of znT[t][CB..CB+15])
    const _Float16* gsrc = zn_tr + (size_t)tid * TWON + col0;
    unsigned ldsA[3];
    #pragma unroll
    for (int b = 0; b < 3; ++b)
        ldsA[b] = (unsigned)(uintptr_t)&panel[b][tid * KSTRIDE];

    // prologue: tiles 0 and 1 in flight (2-tile prefetch lead)
    panel_fill_async(gsrc,      ldsA[0]);
    panel_fill_async(gsrc + 16, ldsA[1]);

    int cur = 0;                      // ti % 3, maintained by rotation
    for (int ti = 0; ti < NTILES; ++ti) {
        const int CB = col0 + ti * 16;

        // own-thread fills complete in order: cnt<=2 leaves only tile ti+1
        if (ti + 1 < NTILES) asm volatile("s_wait_asynccnt 0x2" ::: "memory");
        else                 asm volatile("s_wait_asynccnt 0x0" ::: "memory");
        __syncthreads();  // tile ti fully in LDS; buffer (ti+2)%3 free to refill

        if (ti + 2 < NTILES) {
            const int nxt = (cur + 2 >= 3) ? cur - 1 : cur + 2;  // (ti+2)%3
            panel_fill_async(gsrc + (ti + 2) * 16, ldsA[nxt]);
        }

        // ---- pull all 8 B fragments from LDS, then 8 back-to-back WMMAs ---
        const _Float16* cbuf = panel[cur];
        v16h b[8];
        #pragma unroll
        for (int kk = 0; kk < 8; ++kk) {
            #pragma unroll
            for (int s = 0; s < 16; ++s)
                b[kk][s] = cbuf[(kk * 32 + (s < 8 ? 8 * hi + s : 16 + 8 * hi + (s - 8))) * KSTRIDE + m];
        }

        v8f c = {};
        #pragma unroll
        for (int kk = 0; kk < 8; ++kk)
            c = __builtin_amdgcn_wmma_f32_16x16x32_f16(
                    false, a[kk], false, b[kk], (short)0, c, false, false);

        // C layout: VGPR v -> row R + v + 8*hi ; lane&15 -> col CB + m.
        #pragma unroll
        for (int v = 0; v < 8; ++v) {
            const int grow = R + v + 8 * hi;
            const int gcol = CB + m;
            const float val = c[v] * INV_T;
            if (grow != gcol)           fsum[v] += __expf(val); // mask diagonal
            if (gcol == (grow ^ NPAIR)) fpos[v] = val;          // label = row^N
        }

        cur = (cur == 2) ? 0 : cur + 1;
    }

    // ---- reduce across the 16 lanes sharing each row (xor butterfly) ------
    #pragma unroll
    for (int v = 0; v < 8; ++v) {
        float s = fsum[v], p = fpos[v];
        #pragma unroll
        for (int mk = 1; mk < 16; mk <<= 1) {
            s += __shfl_xor(s, mk, 32);
            p += __shfl_xor(p, mk, 32);
        }
        if (m == 0) {
            const int grow = R + v + 8 * hi;
            atomicAdd(&S[grow], s);
            atomicAdd(&P[grow], p);
        }
    }
}

// --- kernel 3: loss_r = log(S_r) - P_r ; mean over 2N rows -----------------
__global__ void nt_finalize(const float* __restrict__ S,
                            const float* __restrict__ P,
                            float* __restrict__ out) {
    const int idx = blockIdx.x * blockDim.x + threadIdx.x;
    float l = __logf(S[idx]) - P[idx];

    __shared__ float red[256];
    red[threadIdx.x] = l;
    __syncthreads();
    #pragma unroll
    for (int s = 128; s > 0; s >>= 1) {
        if (threadIdx.x < s) red[threadIdx.x] += red[threadIdx.x + s];
        __syncthreads();
    }
    if (threadIdx.x == 0) atomicAdd(out, red[0] * (1.0f / (float)TWON));
}

// ---------------------------------------------------------------------------
extern "C" void kernel_launch(void* const* d_in, const int* in_sizes, int n_in,
                              void* d_out, int out_size, void* d_ws, size_t ws_size,
                              hipStream_t stream) {
    const float* zi = (const float*)d_in[0];
    const float* zj = (const float*)d_in[1];
    float* out = (float*)d_out;

    // workspace layout: zn_rm (4MB f16) | zn_tr (4MB f16) | S (32KB) | P (32KB)
    _Float16* zn_rm = (_Float16*)d_ws;
    _Float16* zn_tr = zn_rm + (size_t)TWON * D;
    float*    S     = (float*)(zn_tr + (size_t)TWON * D);
    float*    P     = S + TWON;

    nt_init<<<TWON / 256, 256, 0, stream>>>(S, P, out);
    nt_normalize<<<TWON, D, 0, stream>>>(zi, zj, zn_rm, zn_tr);
    nt_simloss<<<dim3(TWON / (8 * 16), TWON / COLSLICE, 1), 256, 0, stream>>>(
        zn_rm, zn_tr, S, P);
    nt_finalize<<<TWON / 256, 256, 0, stream>>>(S, P, out);
}
